// alignment_cnn_47502338294138
// MI455X (gfx1250) — matrix operand order repaired and checked by reading, not compiled, not run on hardware
//
#include <hip/hip_runtime.h>
#include <hip/hip_fp16.h>
#include <math.h>

typedef __attribute__((ext_vector_type(16))) _Float16 v16h;
typedef __attribute__((ext_vector_type(8)))  float    v8f;

#define HD   128   // hidden dim
#define LW   64    // words per doc
#define KN   100   // kernel_num
#define KPAD 112   // padded to 7 WMMA tiles
#define WLD  256   // packed-weight row length (halves)

// LDS strides (in elements)
#define SQ   136   // half:  Q/A      [64][128]
#define SQT  72    // half:  QT/AT    [128][64]
#define SS   72    // float: score    [64][64]
#define SW   72    // half:  wq/wa    [64][64]
#define ST   264   // half:  tq/ta    [64][256]
#define SC   113   // float: conv out [64][112]

// dynamic-LDS byte offsets.
// Overlays: [phase 1] sIdx at OFF_TA; [phases 2-3] sS at OFF_TQ (sTq/sTa not
// written until phase 4); [phases 5+] sC*/sPool over region X (sQ..sWa dead).
#define OFF_TQ   0                       // 33792  (sTq; sS overlay)
#define OFF_TA   33792                   // 33792  (sTa; sIdx overlay)
#define OFF_X    67584
#define OFF_Q    (OFF_X)                 // 17408
#define OFF_A    (OFF_Q  + 17408)        // 17408
#define OFF_QT   (OFF_A  + 17408)        // 18432
#define OFF_AT   (OFF_QT + 18432)        // 18432
#define OFF_WQ   (OFF_AT + 18432)        //  9216
#define OFF_WA   (OFF_WQ + 9216)         //  9216
#define SMEM_BYTES (OFF_WA + 9216)       // 157696 (< 160KB -> 2 WGs per WGP)
// phase-5+ overlay of region X (3*28928 + 1600 = 88384 <= 90112)
#define OFF_C1   (OFF_X)
#define OFF_C2A  (OFF_C1  + 28928)       // 64*113*4 = 28928
#define OFF_C2B  (OFF_C2A + 28928)
#define OFF_POOL (OFF_C2B + 28928)       // 400 floats

#define WPACK_BYTES (3 * KPAD * WLD * 2) // 172032

// ---- WMMA fragment helpers (gfx1250 wave32; A/B in contiguous k order, D per ISA layout) ----

// A operand (16x32 f16): lane<16 -> row M=lane, K=k0+{0..15};
// lane>=16 -> row M=lane-16, K=k0+{16..31}. Row-major source.
static __device__ inline v16h frag_a_rowmajor(const _Float16* __restrict__ src,
                                              int stride, int m0, int k0) {
  int lane = threadIdx.x & 31;
  const _Float16* p = src + (size_t)(m0 + (lane & 15)) * stride + k0 + ((lane >> 4) << 4);
  v16h r;
#pragma unroll
  for (int e = 0; e < 16; ++e) r[e] = p[e];
  return r;
}

// B operand (32x16 f16): lane -> column N=n0+(lane&15); element e holds
// K = k0 + (lane>=16?16:0) + e.  Source stored transposed (T[n][k] row-major).
static __device__ inline v16h frag_b_transposed(const _Float16* __restrict__ srcT,
                                                int stride, int n0, int k0) {
  int lane = threadIdx.x & 31;
  const _Float16* p = srcT + (size_t)(n0 + (lane & 15)) * stride + k0 + ((lane >> 4) << 4);
  v16h r;
#pragma unroll
  for (int e = 0; e < 16; ++e) r[e] = p[e];
  return r;
}

// B operand from prepacked f16 weights [KPAD][WLD] (zero-padded rows).
static __device__ inline v16h frag_b_packed(const _Float16* __restrict__ Wp,
                                            int n0, int k0) {
  int lane = threadIdx.x & 31;
  const _Float16* p = Wp + (size_t)(n0 + (lane & 15)) * WLD + k0 + ((lane >> 4) << 4);
  v16h r;
#pragma unroll
  for (int e = 0; e < 16; ++e) r[e] = p[e];
  return r;
}

// B operand from row-major f32 weights (fallback when no workspace).
static __device__ inline v16h frag_b_weight(const float* __restrict__ W, int ldw,
                                            int nrows, int n0, int k0) {
  int lane = threadIdx.x & 31;
  int n = n0 + (lane & 15);
  const float* p = W + (size_t)n * ldw + k0 + ((lane >> 4) << 4);
  bool ok = (n < nrows);
  v16h r;
#pragma unroll
  for (int e = 0; e < 16; ++e) r[e] = ok ? (_Float16)p[e] : (_Float16)0.f;
  return r;
}

// D/C (16x16 f32): lane<16 -> N=lane, VGPR r -> M=r; lane>=16 -> N=lane-16, M=r+8.
static __device__ inline void store_d(float* __restrict__ dst, int stride,
                                      int m0, int n0, v8f d) {
  int lane = threadIdx.x & 31;
  int n  = n0 + (lane & 15);
  int mb = m0 + ((lane >> 4) << 3);
#pragma unroll
  for (int r = 0; r < 8; ++r) dst[(size_t)(mb + r) * stride + n] = d[r];
}

static __device__ inline v8f wmma_f16(v16h a, v16h b, v8f c) {
  return __builtin_amdgcn_wmma_f32_16x16x32_f16(false, a, false, b, (short)0, c,
                                                false, false);
}

// ---- prep: pack W1 / W2[:,0,:] / W2[:,1,:] into zero-padded f16 [3][112][256]
__global__ __launch_bounds__(256) void
pack_weights_kernel(const float* __restrict__ W1, const float* __restrict__ W2,
                    __half* __restrict__ wp) {
  int i = blockIdx.x * 256 + threadIdx.x;   // 3*112*256 = 86016 total
  if (i >= 3 * KPAD * WLD) return;
  int g = i / (KPAD * WLD);
  int r = (i / WLD) % KPAD;
  int d = i & (WLD - 1);
  float v = 0.f;
  if (r < KN) {
    if (g == 0)      v = W1[(size_t)r * 256 + d];
    else if (g == 1) v = W2[(size_t)r * 512 + d];
    else             v = W2[(size_t)r * 512 + 256 + d];
  }
  wp[i] = __float2half(v);
}

template <bool PACKED>
__global__ __launch_bounds__(256) void
alignment_cnn_fused_kernel(const float* __restrict__ emb,
                           const int*   __restrict__ q_idx,
                           const int*   __restrict__ a_idx,
                           const float* __restrict__ W1, const float* __restrict__ b1,
                           const float* __restrict__ W2, const float* __restrict__ b2,
                           const float* __restrict__ Wd, const float* __restrict__ bd,
                           const __half* __restrict__ wpackh,
                           float* __restrict__ out) {
  extern __shared__ char smem[];
  _Float16* sQ   = (_Float16*)(smem + OFF_Q);
  _Float16* sA   = (_Float16*)(smem + OFF_A);
  _Float16* sQT  = (_Float16*)(smem + OFF_QT);
  _Float16* sAT  = (_Float16*)(smem + OFF_AT);
  _Float16* sWq  = (_Float16*)(smem + OFF_WQ);
  _Float16* sWa  = (_Float16*)(smem + OFF_WA);
  float*    sS   = (float*)(smem + OFF_TQ);     // overlay (phases 2-3 only)
  int*      sIdx = (int*)(smem + OFF_TA);       // overlay (phase 1 only)
  _Float16* sTq  = (_Float16*)(smem + OFF_TQ);
  _Float16* sTa  = (_Float16*)(smem + OFF_TA);
  float*    sC1  = (float*)(smem + OFF_C1);
  float*    sC2a = (float*)(smem + OFF_C2A);
  float*    sC2b = (float*)(smem + OFF_C2B);
  float*    sPool= (float*)(smem + OFF_POOL);
  const _Float16* wpack = (const _Float16*)wpackh;

  const int tid  = threadIdx.x;
  const int wave = tid >> 5;
  const int bidx = blockIdx.x;

  // ---- Phase 0: stage indices
  if (tid < LW)               sIdx[tid]      = q_idx[(size_t)bidx * LW + tid];
  else if (tid < 2 * LW)      sIdx[tid]      = a_idx[(size_t)bidx * LW + (tid - LW)];
  __syncthreads();

  // ---- Phase 1: gather embeddings, fp32 -> f16, row-major + transposed copies
  for (int i = tid; i < LW * HD; i += 256) {
    int row = i >> 7, col = i & (HD - 1);
    float xq = emb[(size_t)sIdx[row] * HD + col];
    float xa = emb[(size_t)sIdx[LW + row] * HD + col];
    _Float16 hq = (_Float16)xq, ha = (_Float16)xa;
    sQ [row * SQ  + col] = hq;  sQT[col * SQT + row] = hq;
    sA [row * SQ  + col] = ha;  sAT[col * SQT + row] = ha;
  }
  __syncthreads();

  // ---- Phase 2: score = Q @ A^T  (64x64, K=128) -> sS (f32)
  for (int t = wave; t < 16; t += 8) {
    int mi = t >> 2, ni = t & 3;
    v8f acc = {};
#pragma unroll
    for (int kt = 0; kt < 4; ++kt) {
      v16h a = frag_a_rowmajor(sQ, SQ, mi * 16, kt * 32);
      v16h b = frag_b_transposed(sA, SQ, ni * 16, kt * 32);  // Mat2[k][n]=A[n][k]
      acc = wmma_f16(a, b, acc);
    }
    store_d(sS, SS, mi * 16, ni * 16, acc);
  }
  __syncthreads();

  // ---- Phase 3: wq = softmax over a (rows); wa = softmax over q (cols)
  if (tid < LW) {
    int r = tid;
    float mx = -1e30f;
    for (int j = 0; j < LW; ++j) mx = fmaxf(mx, sS[r * SS + j]);
    float sum = 0.f;
    for (int j = 0; j < LW; ++j) sum += expf(sS[r * SS + j] - mx);
    float inv = 1.f / sum;
    for (int j = 0; j < LW; ++j)
      sWq[r * SW + j] = (_Float16)(expf(sS[r * SS + j] - mx) * inv);
  } else if (tid < 2 * LW) {
    int a = tid - LW;
    float mx = -1e30f;
    for (int j = 0; j < LW; ++j) mx = fmaxf(mx, sS[j * SS + a]);
    float sum = 0.f;
    for (int j = 0; j < LW; ++j) sum += expf(sS[j * SS + a] - mx);
    float inv = 1.f / sum;
    for (int j = 0; j < LW; ++j)
      sWa[a * SW + j] = (_Float16)(expf(sS[j * SS + a] - mx) * inv);
  }
  __syncthreads();

  // ---- Phase 4: EQ = wq @ A, EA = wa @ Q (64x128, K=64); fuse tq/ta features
  for (int t = wave; t < 64; t += 8) {
    bool isQ = t < 32;
    int tt = isQ ? t : t - 32;
    int mi = tt >> 3, hi = tt & 7;
    const _Float16* sWm = isQ ? sWq : sWa;
    const _Float16* sBT = isQ ? sAT : sQT;   // Mat2[k][n] = (A|Q)[k][n]
    v8f acc = {};
#pragma unroll
    for (int kt = 0; kt < 2; ++kt) {
      v16h a = frag_a_rowmajor(sWm, SW, mi * 16, kt * 32);
      v16h b = frag_b_transposed(sBT, SQT, hi * 16, kt * 32);
      acc = wmma_f16(a, b, acc);
    }
    const _Float16* sX  = isQ ? sQ  : sA;
    _Float16*       sTo = isQ ? sTq : sTa;
    int lane = tid & 31;
    int h  = hi * 16 + (lane & 15);
    int mb = mi * 16 + ((lane >> 4) << 3);
#pragma unroll
    for (int r = 0; r < 8; ++r) {
      int m = mb + r;
      float x = (float)sX[m * SQ + h];
      float e = acc[r];
      float d = x - e;
      sTo[m * ST + h]      = (_Float16)(d * d);   // (X - EX)^2
      sTo[m * ST + HD + h] = (_Float16)(x * e);   // X * EX
    }
  }
  __syncthreads();

  // ---- Phases 5-7: conv-as-GEMM per branch (q then a), pool into sPool[400]
  for (int br = 0; br < 2; ++br) {
    const _Float16* sT = br ? sTa : sTq;
    for (int t = wave; t < 84; t += 8) {
      int g = t / 28, rr = t % 28;
      int mi = rr / 7, ni = rr % 7;
      v8f acc = {};
#pragma unroll
      for (int kt = 0; kt < 8; ++kt) {
        int k0 = kt * 32;
        v16h a = frag_a_rowmajor(sT, ST, mi * 16, k0);
        v16h b;
        if (PACKED) {
          b = frag_b_packed(wpack + (size_t)g * KPAD * WLD, ni * 16, k0);
        } else {
          if (g == 0)      b = frag_b_weight(W1,       256, KN, ni * 16, k0);
          else if (g == 1) b = frag_b_weight(W2,       512, KN, ni * 16, k0);
          else             b = frag_b_weight(W2 + 256, 512, KN, ni * 16, k0);
        }
        acc = wmma_f16(a, b, acc);
      }
      float* dst = (g == 0) ? sC1 : ((g == 1) ? sC2a : sC2b);
      store_d(dst, SC, mi * 16, ni * 16, acc);
    }
    __syncthreads();
    // pooling: relu + max over window positions
    if (tid < KPAD) {
      int k = tid;
      float bias = (k < KN) ? b1[k] : 0.f;
      float m = 0.f;
      for (int l = 0; l < LW; ++l) {
        float v = sC1[l * SC + k] + bias;
        v = v > 0.f ? v : 0.f;
        m = v > m ? v : m;
      }
      if (k < KN) sPool[br * 200 + k] = m;
    } else if (tid < 2 * KPAD) {
      int k = tid - KPAD;
      float bias = (k < KN) ? b2[k] : 0.f;
      float m = 0.f;
      for (int l = 0; l < LW - 1; ++l) {
        float v = sC2a[l * SC + k] + sC2b[(l + 1) * SC + k] + bias;
        v = v > 0.f ? v : 0.f;
        m = v > m ? v : m;
      }
      if (k < KN) sPool[br * 200 + 100 + k] = m;
    }
    __syncthreads();
  }

  // ---- Phase 8: final linear (400 -> 2) + log_softmax, wave-parallel
  if (wave == 0) {
    int lane = tid;
    float p0 = 0.f, p1 = 0.f;
    for (int j = lane; j < 400; j += 32) {
      float p = sPool[j];
      p0 += p * Wd[j];
      p1 += p * Wd[400 + j];
    }
#pragma unroll
    for (int off = 16; off > 0; off >>= 1) {
      p0 += __shfl_xor(p0, off, 32);
      p1 += __shfl_xor(p1, off, 32);
    }
    if (lane == 0) {
      float s0 = p0 + bd[0], s1 = p1 + bd[1];
      float mx  = fmaxf(s0, s1);
      float lse = mx + logf(expf(s0 - mx) + expf(s1 - mx));
      out[(size_t)bidx * 2 + 0] = s0 - lse;
      out[(size_t)bidx * 2 + 1] = s1 - lse;
    }
  }
}

extern "C" void kernel_launch(void* const* d_in, const int* in_sizes, int n_in,
                              void* d_out, int out_size, void* d_ws, size_t ws_size,
                              hipStream_t stream) {
  (void)in_sizes; (void)n_in; (void)out_size;
  const float* emb = (const float*)d_in[0];
  const int*   qI  = (const int*)d_in[1];
  const int*   aI  = (const int*)d_in[2];
  const float* W1  = (const float*)d_in[3];
  const float* b1  = (const float*)d_in[4];
  const float* W2  = (const float*)d_in[5];
  const float* b2  = (const float*)d_in[6];
  const float* Wd  = (const float*)d_in[7];
  const float* bd  = (const float*)d_in[8];
  float* out = (float*)d_out;

  bool packed = (d_ws != nullptr) && (ws_size >= (size_t)WPACK_BYTES);

  if (packed) {
    __half* wp = (__half*)d_ws;
    pack_weights_kernel<<<dim3((3 * KPAD * WLD + 255) / 256), dim3(256), 0, stream>>>(
        W1, W2, wp);
    (void)hipFuncSetAttribute(
        reinterpret_cast<const void*>(&alignment_cnn_fused_kernel<true>),
        hipFuncAttributeMaxDynamicSharedMemorySize, SMEM_BYTES);
    alignment_cnn_fused_kernel<true><<<dim3(1024), dim3(256), SMEM_BYTES, stream>>>(
        emb, qI, aI, W1, b1, W2, b2, Wd, bd, (const __half*)wp, out);
  } else {
    (void)hipFuncSetAttribute(
        reinterpret_cast<const void*>(&alignment_cnn_fused_kernel<false>),
        hipFuncAttributeMaxDynamicSharedMemorySize, SMEM_BYTES);
    alignment_cnn_fused_kernel<false><<<dim3(1024), dim3(256), SMEM_BYTES, stream>>>(
        emb, qI, aI, W1, b1, W2, b2, Wd, bd, nullptr, out);
  }
}
